// PowerMethod_2645699855023
// MI455X (gfx1250) — matrix operand order repaired and checked by reading, not compiled, not run on hardware
//
#include <hip/hip_runtime.h>
#include <hip/hip_bf16.h>
#include <math.h>

// ---------------------------------------------------------------------------
// PowerMethod on AhA = A^H A, B=256, M=N=256, 30 iterations.
// Phase 1: batched complex GEMM (34 GFLOP) -> bf16 WMMA, f32 accumulate,
//          fragment-major LDS staging so frag loads are ds_load_b128.
// Phase 2: per-batch power iteration; G tile (256 rows x 1024B, interleaved
//          re/im bf16) staged into LDS by the Tensor Data Mover with
//          hardware row padding, then 30 matvec iterations from LDS.
// Workspace: G interleaved complex bf16: B*N*N uint = 64 MB.
// ---------------------------------------------------------------------------

typedef __attribute__((ext_vector_type(16))) __bf16 v16bf;
typedef __attribute__((ext_vector_type(8)))  __bf16 v8bf;
typedef __attribute__((ext_vector_type(8)))  float  v8f;
typedef __attribute__((ext_vector_type(4)))  unsigned int v4u;
typedef __attribute__((ext_vector_type(4)))  int    v4i;
typedef __attribute__((ext_vector_type(8)))  int    v8i;

#define BATCH 256
#define NDIM  256
#define MDIM  256
#define LPAD  24          // per-lane fragment slot (16 used + 8 pad) in bf16
#define GROW  520         // kernel-2 LDS row stride in bf16 (512 + 8 pad)

typedef unsigned short bf16raw;

__device__ __forceinline__ bf16raw f2bf(float f) {
    __bf16 h = (__bf16)f;
    return __builtin_bit_cast(bf16raw, h);
}
__device__ __forceinline__ unsigned int packbf2(float a, float b) {
    return (unsigned int)f2bf(a) | ((unsigned int)f2bf(b) << 16);
}
__device__ __forceinline__ v16bf load_frag(const __bf16* p) {
    v8bf lo = *(const v8bf*)(p);
    v8bf hi = *(const v8bf*)(p + 8);
    return __builtin_shufflevector(lo, hi, 0,1,2,3,4,5,6,7,8,9,10,11,12,13,14,15);
}

// ---------------------------------------------------------------------------
// Kernel 1: G = A^H A (per batch). Grid: BATCH*2(n:128)*4(k:64) = 2048 blocks.
// LDS holds A/B operands already permuted into WMMA fragment order:
//   A (16x32 bf16): lane = (n&15) + 16*((m>>3)&1), elem = (m&7) + ((m>>4)<<3)
//   B (32x16 bf16): lane = (k&15) + 16*((m>>3)&1), elem = (m&7) + ((m>>4)<<3)
// so each lane's fragment is 16 contiguous bf16 -> 2x ds_load_b128.
// Re(G) = Re^T Re + Im^T Im ; Im(G) = Re^T Im - Im^T Re (3 accumulators).
// ---------------------------------------------------------------------------
__global__ __launch_bounds__(256) void ahat_gemm_kernel(
    const float* __restrict__ A_re, const float* __restrict__ A_im,
    unsigned int* __restrict__ G)   // interleaved complex bf16 per element
{
    __shared__ __bf16 lds_a_re[8 * 32 * LPAD];
    __shared__ __bf16 lds_a_im[8 * 32 * LPAD];
    __shared__ __bf16 lds_b_re[4 * 32 * LPAD];
    __shared__ __bf16 lds_b_im[4 * 32 * LPAD];

    const int tid  = threadIdx.x;
    const int b    = blockIdx.x >> 3;
    const int tile = blockIdx.x & 7;
    const int n0   = (tile >> 2) * 128;
    const int k0   = (tile & 3) * 64;

    const int wave = tid >> 5;
    const int lane = tid & 31;
    const int half = lane >> 4;
    const int mrow = lane & 15;
    const int wrow = wave * 16;

    const size_t abase = (size_t)b * (MDIM * NDIM);
    const float* Ar = A_re + abase;
    const float* Ai = A_im + abase;

    // Staging task decomposition (per 32-row M chunk):
    //  n-side: 8 m-groups x 32 n-groups of 4x4 blocks -> one block/thread.
    const int a_mg = tid >> 5, a_ng = tid & 31;
    const int a_m0 = a_mg * 4, a_nn = a_ng * 4;
    const int a_half = (a_m0 >> 3) & 1;
    const int a_e0 = (a_m0 & 7) + ((a_m0 >> 4) << 3);
    //  k-side: 8 m-groups x 16 k-groups x 2 planes -> one task/thread.
    const int b_plane = tid >> 7;
    const int b_blk = tid & 127;
    const int b_m0 = (b_blk >> 4) * 4, b_kk = (b_blk & 15) * 4;
    const int b_half = (b_m0 >> 3) & 1;
    const int b_e0 = (b_m0 & 7) + ((b_m0 >> 4) << 3);

    v8f acc_re[4], acc_ri[4], acc_ir[4];
#pragma unroll
    for (int s = 0; s < 4; ++s) { acc_re[s] = {}; acc_ri[s] = {}; acc_ir[s] = {}; }

    for (int mc = 0; mc < MDIM; mc += 32) {
        // ---- stage n-side (both planes), 4x4 block transpose in registers
        {
            const float* pr = Ar + (size_t)(mc + a_m0) * NDIM + n0 + a_nn;
            const float* pi = Ai + (size_t)(mc + a_m0) * NDIM + n0 + a_nn;
            float4 r0 = *(const float4*)(pr);
            float4 r1 = *(const float4*)(pr + NDIM);
            float4 r2 = *(const float4*)(pr + 2 * NDIM);
            float4 r3 = *(const float4*)(pr + 3 * NDIM);
            float4 i0 = *(const float4*)(pi);
            float4 i1 = *(const float4*)(pi + NDIM);
            float4 i2 = *(const float4*)(pi + 2 * NDIM);
            float4 i3 = *(const float4*)(pi + 3 * NDIM);
#pragma unroll
            for (int i = 0; i < 4; ++i) {
                int n = a_nn + i;
                int slot = (((n >> 4) * 32) + (n & 15) + 16 * a_half) * LPAD + a_e0;
                uint2 vr; uint2 vi;
                vr.x = packbf2((&r0.x)[i], (&r1.x)[i]);
                vr.y = packbf2((&r2.x)[i], (&r3.x)[i]);
                vi.x = packbf2((&i0.x)[i], (&i1.x)[i]);
                vi.y = packbf2((&i2.x)[i], (&i3.x)[i]);
                *(uint2*)&lds_a_re[slot] = vr;
                *(uint2*)&lds_a_im[slot] = vi;
            }
        }
        // ---- stage k-side (one plane per thread-half)
        {
            const float* p = (b_plane ? Ai : Ar) + (size_t)(mc + b_m0) * NDIM + k0 + b_kk;
            __bf16* ldsb = b_plane ? lds_b_im : lds_b_re;
            float4 r0 = *(const float4*)(p);
            float4 r1 = *(const float4*)(p + NDIM);
            float4 r2 = *(const float4*)(p + 2 * NDIM);
            float4 r3 = *(const float4*)(p + 3 * NDIM);
#pragma unroll
            for (int i = 0; i < 4; ++i) {
                int k = b_kk + i;
                int slot = (((k >> 4) * 32) + (k & 15) + 16 * b_half) * LPAD + b_e0;
                uint2 v;
                v.x = packbf2((&r0.x)[i], (&r1.x)[i]);
                v.y = packbf2((&r2.x)[i], (&r3.x)[i]);
                *(uint2*)&ldsb[slot] = v;
            }
        }
        __syncthreads();

        // ---- fragments: contiguous per-lane, 2x b128 each
        v16bf a_re = load_frag(&lds_a_re[(wave * 32 + lane) * LPAD]);
        v16bf a_im = load_frag(&lds_a_im[(wave * 32 + lane) * LPAD]);
#pragma unroll
        for (int s = 0; s < 4; ++s) {
            v16bf b_re = load_frag(&lds_b_re[(s * 32 + lane) * LPAD]);
            v16bf b_im = load_frag(&lds_b_im[(s * 32 + lane) * LPAD]);
            acc_re[s] = __builtin_amdgcn_wmma_f32_16x16x32_bf16(
                false, a_re, false, b_re, (short)0, acc_re[s], false, false);
            acc_re[s] = __builtin_amdgcn_wmma_f32_16x16x32_bf16(
                false, a_im, false, b_im, (short)0, acc_re[s], false, false);
            acc_ri[s] = __builtin_amdgcn_wmma_f32_16x16x32_bf16(
                false, a_re, false, b_im, (short)0, acc_ri[s], false, false);
            acc_ir[s] = __builtin_amdgcn_wmma_f32_16x16x32_bf16(
                false, a_im, false, b_re, (short)0, acc_ir[s], false, false);
        }
        __syncthreads();
    }

    // Epilogue: C/D layout row M = r + 8*half, col N = lane&15.
    // Store interleaved complex bf16 packed in one b32 per element.
    const size_t gbase = (size_t)b * (NDIM * NDIM);
#pragma unroll
    for (int s = 0; s < 4; ++s) {
        v8f ore = acc_re[s];
        v8f oim = acc_ri[s] - acc_ir[s];
#pragma unroll
        for (int r = 0; r < 8; ++r) {
            int n = n0 + wrow + half * 8 + r;
            int k = k0 + s * 16 + mrow;
            G[gbase + (size_t)n * NDIM + k] = packbf2(ore[r], oim[r]);
        }
    }
}

// ---------------------------------------------------------------------------
// Kernel 2: power iteration. One block per batch. The batch's G tile
// (256 rows x 1024 B) is staged into LDS by the Tensor Data Mover with
// hardware padding: pad_interval=7 (every 256 dwords) pad_amount=3 (4 dwords)
// -> LDS row stride 1040 B = GROW halfs (16B-aligned, bank-skewed).
// ---------------------------------------------------------------------------
__global__ __launch_bounds__(256) void power_iter_kernel(
    const float* __restrict__ v_re_in, const float* __restrict__ v_im_in,
    const unsigned int* __restrict__ G,
    float* __restrict__ out, const int* __restrict__ n_iter_p)
{
    extern __shared__ __bf16 lds_g[];          // [256][GROW] interleaved re,im
    __shared__ float vs_re[NDIM];
    __shared__ float vs_im[NDIM];
    __shared__ float red[8];

    const int b   = blockIdx.x;
    const int tid = threadIdx.x;
    const int iters = *n_iter_p;

    // ---- TDM: one wave issues the tensor DMA descriptor
    if (tid < 32) {
        unsigned int lds_off = (unsigned int)(uintptr_t)(void*)lds_g; // LDS byte offset
        unsigned long long ga =
            (unsigned long long)(uintptr_t)(const void*)(G + (size_t)b * NDIM * NDIM);

        v4u g0;
        g0.x = 1u;                                        // count=1, user desc
        g0.y = lds_off;                                   // lds_addr
        g0.z = (unsigned int)(ga & 0xffffffffu);          // global_addr[31:0]
        g0.w = (unsigned int)((ga >> 32) & 0x01ffffffu)   // global_addr[56:32]
               | (2u << 30);                              // type = 2 (image)

        v8i g1;
        g1[0] = (2 << 16)            // data_size = 4 bytes
              | (1 << 20)            // pad_enable
              | (7 << 22)            // pad_interval: 256 dwords
              | (3 << 25);           // pad_amount:   4 dwords
        g1[1] = (int)(256u << 16);   // tensor_dim0 = 256 (low 16 in [63:48])
        g1[2] = (int)(256u << 16);   // tensor_dim1 = 256 (low 16 in [95:80])
        g1[3] = (int)(256u << 16);   // tile_dim0 = 256   ([127:112])
        g1[4] = 256;                 // tile_dim1 = 256   ([143:128]); tile_dim2=0
        g1[5] = 256;                 // tensor_dim0_stride = 256 ([191:160])
        g1[6] = (int)(256u << 16);   // tensor_dim1_stride low16 ([223:208])
        g1[7] = 0;

        v4i g2 = (v4i)0;
        v4i g3 = (v4i)0;
#if defined(__clang_major__) && (__clang_major__ >= 23)
        v8i g4 = (v8i)0;
        __builtin_amdgcn_tensor_load_to_lds(g0, g1, g2, g3, g4, 0);
#else
        __builtin_amdgcn_tensor_load_to_lds(g0, g1, g2, g3, 0);
#endif
        __builtin_amdgcn_s_wait_tensorcnt(0);
    }
    vs_re[tid] = v_re_in[b * NDIM + tid];
    vs_im[tid] = v_im_in[b * NDIM + tid];
    __syncthreads();   // publishes TDM-filled LDS to all waves

    float eig = 0.0f;
    for (int it = 0; it < iters; ++it) {
        // w[n] = sum_k G[n][k] * v[k]  (complex), row n = tid
        float wr = 0.0f, wi = 0.0f;
        const __bf16* row = lds_g + tid * GROW;
#pragma unroll 2
        for (int kq = 0; kq < NDIM; kq += 4) {
            v8bf p = *(const v8bf*)(row + 2 * kq);   // 4 complex, ds_load_b128
#pragma unroll
            for (int j = 0; j < 4; ++j) {
                float gr = (float)p[2 * j];
                float gi = (float)p[2 * j + 1];
                float vr = vs_re[kq + j], vi = vs_im[kq + j];
                wr = fmaf(gr, vr, wr);
                wr = fmaf(-gi, vi, wr);
                wi = fmaf(gr, vi, wi);
                wi = fmaf(gi, vr, wi);
            }
        }

        // ||w||^2: wave32 shuffle reduce, then cross-wave via LDS.
        float s = wr * wr + wi * wi;
#pragma unroll
        for (int o = 16; o > 0; o >>= 1) s += __shfl_down(s, o, 32);
        if ((tid & 31) == 0) red[tid >> 5] = s;
        __syncthreads();

        float tot = 0.0f;
#pragma unroll
        for (int wv = 0; wv < 8; ++wv) tot += red[wv];
        eig = sqrtf(tot);
        float inv = 1.0f / (eig + 1e-6f);
        vs_re[tid] = wr * inv;
        vs_im[tid] = wi * inv;
        __syncthreads();
    }

    if (tid == 0) out[b] = eig;
}

// ---------------------------------------------------------------------------
extern "C" void kernel_launch(void* const* d_in, const int* in_sizes, int n_in,
                              void* d_out, int out_size, void* d_ws, size_t ws_size,
                              hipStream_t stream) {
    const float* A_re = (const float*)d_in[0];
    const float* A_im = (const float*)d_in[1];
    const float* v_re = (const float*)d_in[2];
    const float* v_im = (const float*)d_in[3];
    const int*   n_it = (const int*)d_in[4];

    unsigned int* G = (unsigned int*)d_ws;   // B*N*N packed complex bf16, 64 MB

    ahat_gemm_kernel<<<dim3(BATCH * 8), dim3(256), 0, stream>>>(A_re, A_im, G);

    size_t dyn = (size_t)NDIM * GROW * sizeof(bf16raw);   // 260 KB
    hipFuncSetAttribute((const void*)power_iter_kernel,
                        hipFuncAttributeMaxDynamicSharedMemorySize, (int)dyn);
    power_iter_kernel<<<dim3(BATCH), dim3(256), dyn, stream>>>(
        v_re, v_im, G, (float*)d_out, n_it);
}
